// GCNPredictor_5497558139599
// MI455X (gfx1250) — matrix operand order repaired and checked by reading, not compiled, not run on hardware
//
#include <hip/hip_runtime.h>
#include <hip/hip_bf16.h>
#include <math.h>

typedef __attribute__((ext_vector_type(16))) _Float16 v16h;
typedef __attribute__((ext_vector_type(8)))  _Float16 v8h;
typedef __attribute__((ext_vector_type(8)))  float    v8f;

#define BM 128
#define BN 64
#define BK 32
#define LDA (BK + 8)          // 40 halfs = 80B row stride: keeps 16B stores aligned
#define NEG_SLOPE 0.2f

union FragU { v16h v; unsigned u[8]; };

// ---------------------------------------------------------------------------
// Stage one BMxBK A-tile and BKxBN B-tile (transposed) into LDS as f16.
// KG: K-boundary guard (tail step only). AFF/RS: fused BN-affine / row scale.
// ---------------------------------------------------------------------------
template<bool KG, bool AFF, bool RS>
__device__ __forceinline__ void stage_tile(
    const float* __restrict__ A, const float* __restrict__ W,
    const float* __restrict__ affS, const float* __restrict__ affB, float rsv,
    _Float16 (*lsA)[LDA], _Float16 (*lsB)[LDA],
    int grow, int kb, int K, int n0, int N, int t)
{
    // ----- A tile: thread owns 16 consecutive k of one row -----
    const int rA   = t >> 1;
    const int segA = (t & 1) * 16;
    if (!KG) {
        const float* p = A + (long)grow * K + kb + segA;
        float vv[16];
        #pragma unroll
        for (int q = 0; q < 4; ++q) {
            float4 f = *(const float4*)(p + 4 * q);
            vv[4 * q + 0] = f.x; vv[4 * q + 1] = f.y;
            vv[4 * q + 2] = f.z; vv[4 * q + 3] = f.w;
        }
        if (AFF) {
            const float* ps = affS + kb + segA;
            const float* pb = affB + kb + segA;
            #pragma unroll
            for (int q = 0; q < 4; ++q) {
                float4 s = *(const float4*)(ps + 4 * q);
                float4 b = *(const float4*)(pb + 4 * q);
                vv[4 * q + 0] = vv[4 * q + 0] * s.x + b.x;
                vv[4 * q + 1] = vv[4 * q + 1] * s.y + b.y;
                vv[4 * q + 2] = vv[4 * q + 2] * s.z + b.z;
                vv[4 * q + 3] = vv[4 * q + 3] * s.w + b.w;
            }
        }
        if (RS) {
            #pragma unroll
            for (int j = 0; j < 16; ++j) vv[j] *= rsv;
        }
        v8h h0, h1;
        #pragma unroll
        for (int j = 0; j < 8; ++j) { h0[j] = (_Float16)vv[j]; h1[j] = (_Float16)vv[8 + j]; }
        *(v8h*)&lsA[rA][segA]     = h0;
        *(v8h*)&lsA[rA][segA + 8] = h1;
    } else {
        #pragma unroll
        for (int j = 0; j < 16; ++j) {
            int k = kb + segA + j;
            float v = 0.0f;
            if (k < K) {
                v = A[(long)grow * K + k];
                if (AFF) v = v * affS[k] + affB[k];
                if (RS)  v *= rsv;
            }
            lsA[rA][segA + j] = (_Float16)v;
        }
    }

    // ----- B tile: thread owns 8 consecutive n of one k; store transposed -----
    const int kB    = t >> 3;
    const int nsegB = (t & 7) * 8;
    const int gk    = kb + kB;
    const int nb    = n0 + nsegB;
    const bool kin  = !KG || (gk < K);
    if (kin && (nb + 8 <= N)) {
        const float* p = W + (long)gk * N + nb;
        float4 a = *(const float4*)p;
        float4 b = *(const float4*)(p + 4);
        lsB[nsegB + 0][kB] = (_Float16)a.x;
        lsB[nsegB + 1][kB] = (_Float16)a.y;
        lsB[nsegB + 2][kB] = (_Float16)a.z;
        lsB[nsegB + 3][kB] = (_Float16)a.w;
        lsB[nsegB + 4][kB] = (_Float16)b.x;
        lsB[nsegB + 5][kB] = (_Float16)b.y;
        lsB[nsegB + 6][kB] = (_Float16)b.z;
        lsB[nsegB + 7][kB] = (_Float16)b.w;
    } else {
        #pragma unroll
        for (int j = 0; j < 8; ++j) {
            float v = (kin && (nb + j) < N) ? W[(long)gk * N + nb + j] : 0.0f;
            lsB[nsegB + j][kB] = (_Float16)v;
        }
    }
}

// ---------------------------------------------------------------------------
// One BK step of WMMA over the staged tiles: wave owns a 16x64 strip.
// ---------------------------------------------------------------------------
__device__ __forceinline__ void wmma_step(
    _Float16 (*lsA)[LDA], _Float16 (*lsB)[LDA], v8f acc[4], int lane, int wave)
{
    const int hi = lane >> 4;
    const int ml = (lane & 15) + wave * 16;
    FragU fa;
    #pragma unroll
    for (int v = 0; v < 8; ++v) {
        // A 16x32 f16 wave32 layout: VGPR v holds K pair (v<4 ? 2v : 2v+8) + hi*8
        int k0 = (v < 4 ? 2 * v : 2 * v + 8) + hi * 8;
        fa.u[v] = *(const unsigned*)&lsA[ml][k0];
    }
    FragU fb[4];
    #pragma unroll
    for (int tt = 0; tt < 4; ++tt) {
        int nl = (lane & 15) + tt * 16;
        #pragma unroll
        for (int v = 0; v < 8; ++v) {
            // B 32x16 f16 wave32 layout: VGPR v holds K pair (v<4 ? 2v : 2v+8) + hi*8
            int k0 = (v < 4 ? 2 * v : 2 * v + 8) + hi * 8;
            fb[tt].u[v] = *(const unsigned*)&lsB[nl][k0];
        }
    }
    #pragma unroll
    for (int tt = 0; tt < 4; ++tt)
        acc[tt] = __builtin_amdgcn_wmma_f32_16x16x32_f16(
            false, fa.v, false, fb[tt].v, (short)0, acc[tt], false, false);
}

// ---------------------------------------------------------------------------
// Fused WMMA GEMM: out = epilogue( ((A*aff)*rs) @ W + bias ), f32 -> f16 -> f32acc
// ---------------------------------------------------------------------------
template<bool AFF, bool RS, bool LEAKY>
__global__ void __launch_bounds__(256)
gemm_wmma_kernel(const float* __restrict__ A, const float* __restrict__ W,
                 const float* __restrict__ bias,
                 const float* __restrict__ rowScale,
                 const float* __restrict__ affS, const float* __restrict__ affB,
                 float* __restrict__ out, int M, int K, int N)
{
    __shared__ __align__(16) _Float16 lsA[BM][LDA];
    __shared__ __align__(16) _Float16 lsB[BN][LDA];

    const int t    = threadIdx.x;
    const int lane = t & 31;
    const int wave = t >> 5;
    const int row0 = blockIdx.x * BM;
    const int n0   = blockIdx.y * BN;

    v8f acc[4] = {};

    const int grow = row0 + (t >> 1);
    const float rsv = RS ? rowScale[grow] : 1.0f;

    const int kFull = K / BK;
    for (int kc = 0; kc < kFull; ++kc) {
        const int kb = kc * BK;
        if (kc + 1 < kFull)
            __builtin_prefetch(&A[(long)grow * K + kb + BK + (t & 1) * 16], 0, 1);
        stage_tile<false, AFF, RS>(A, W, affS, affB, rsv, lsA, lsB,
                                   grow, kb, K, n0, N, t);
        __syncthreads();
        wmma_step(lsA, lsB, acc, lane, wave);
        __syncthreads();
    }
    if (K % BK) {
        stage_tile<true, AFF, RS>(A, W, affS, affB, rsv, lsA, lsB,
                                  grow, kFull * BK, K, n0, N, t);
        __syncthreads();
        wmma_step(lsA, lsB, acc, lane, wave);
        __syncthreads();
    }

    // epilogue: bias + optional leaky, store f32
    const int hi = lane >> 4;
    const int nl = lane & 15;
    const int rbase = row0 + wave * 16;
    #pragma unroll
    for (int tt = 0; tt < 4; ++tt) {
        int col = n0 + tt * 16 + nl;
        if (col >= N) continue;
        float bv = bias[col];
        #pragma unroll
        for (int v = 0; v < 8; ++v) {
            int row = rbase + v + hi * 8;
            float val = acc[tt][v] + bv;
            if (LEAKY) val = val > 0.0f ? val : NEG_SLOPE * val;
            out[(long)row * N + col] = val;
        }
    }
}

// ---------------------------------------------------------------------------
// rs[row] = sum_j m[row, j]   (one wave32 per row)
// ---------------------------------------------------------------------------
__global__ void rowsum_kernel(const float* __restrict__ m, float* __restrict__ rs,
                              int R, int nrows)
{
    int wid  = (blockIdx.x * blockDim.x + threadIdx.x) >> 5;
    int lane = threadIdx.x & 31;
    if (wid >= nrows) return;
    const float* p = m + (long)wid * R;
    float s = 0.0f;
    for (int j = lane; j < R; j += 32) s += p[j];
    #pragma unroll
    for (int o = 16; o > 0; o >>= 1) s += __shfl_xor(s, o, 32);
    if (lane == 0) rs[wid] = s;
}

// ---------------------------------------------------------------------------
// Per-feature partial sums / sums of squares (coalesced across features)
// ---------------------------------------------------------------------------
__global__ void colstats_kernel(const float* __restrict__ X, float* __restrict__ sum,
                                float* __restrict__ sq, int F, int rowsPerBlock)
{
    int f = threadIdx.x;
    if (f >= F) return;
    long r0 = (long)blockIdx.x * rowsPerBlock;
    float s = 0.0f, s2 = 0.0f;
    for (int i = 0; i < rowsPerBlock; ++i) {
        float v = X[(r0 + i) * F + f];
        s += v; s2 += v * v;
    }
    atomicAdd(&sum[f], s);
    atomicAdd(&sq[f],  s2);
}

__global__ void bn_finalize_kernel(const float* __restrict__ sum, const float* __restrict__ sq,
                                   const float* __restrict__ g, const float* __restrict__ be,
                                   float* __restrict__ scale, float* __restrict__ shift,
                                   int F, float invCnt)
{
    int f = blockIdx.x * blockDim.x + threadIdx.x;
    if (f >= F) return;
    float mu  = sum[f] * invCnt;
    float var = sq[f] * invCnt - mu * mu;
    float sc  = g[f] * rsqrtf(var + 1e-5f);
    scale[f] = sc;
    shift[f] = be[f] - mu * sc;
}

// ---------------------------------------------------------------------------
// T4 = leaky( T3[M,64] @ W4[64,8] + b4 )    one thread per row
// ---------------------------------------------------------------------------
__global__ void fc64x8_kernel(const float* __restrict__ X, const float* __restrict__ W4,
                              const float* __restrict__ b4, float* __restrict__ out, int M)
{
    __shared__ float w[64][8];
    int t = threadIdx.x;
    for (int i = t; i < 64 * 8; i += blockDim.x) w[i >> 3][i & 7] = W4[i];
    __syncthreads();
    int row = blockIdx.x * blockDim.x + t;
    if (row >= M) return;
    float a[8];
    #pragma unroll
    for (int c = 0; c < 8; ++c) a[c] = b4[c];
    for (int k = 0; k < 64; k += 4) {
        float4 x = *(const float4*)&X[(long)row * 64 + k];
        #pragma unroll
        for (int c = 0; c < 8; ++c)
            a[c] += x.x * w[k][c] + x.y * w[k + 1][c] + x.z * w[k + 2][c] + x.w * w[k + 3][c];
    }
    #pragma unroll
    for (int c = 0; c < 8; ++c) {
        float v = a[c];
        out[(long)row * 8 + c] = v > 0.0f ? v : NEG_SLOPE * v;
    }
}

// ---------------------------------------------------------------------------
// Channel BN over [bz, R, 8]: stats over (batch, 8) per roi channel r.
// ---------------------------------------------------------------------------
__global__ void bnch_stats_kernel(const float* __restrict__ X, const float* __restrict__ g3,
                                  const float* __restrict__ be3, float* __restrict__ scale,
                                  float* __restrict__ shift, int bz, int R)
{
    __shared__ float ssum[256], ssq[256];
    int r = blockIdx.x, t = threadIdx.x;
    float s = 0.0f, s2 = 0.0f;
    if (t < bz) {
        const float* p = X + ((long)t * R + r) * 8;
        #pragma unroll
        for (int c = 0; c < 8; ++c) { float v = p[c]; s += v; s2 += v * v; }
    }
    ssum[t] = s; ssq[t] = s2;
    __syncthreads();
    for (int o = 128; o > 0; o >>= 1) {
        if (t < o) { ssum[t] += ssum[t + o]; ssq[t] += ssq[t + o]; }
        __syncthreads();
    }
    if (t == 0) {
        float invC = 1.0f / (float)(bz * 8);
        float mu  = ssum[0] * invC;
        float var = ssq[0] * invC - mu * mu;
        float sc  = g3[r] * rsqrtf(var + 1e-5f);
        scale[r] = sc;
        shift[r] = be3[r] - mu * sc;
    }
}

// apply channel BN, compute sigmoid score; write score to ws and to d_out sc region
__global__ void bnch_apply_score_kernel(const float* __restrict__ X,
                                        const float* __restrict__ scale,
                                        const float* __restrict__ shift,
                                        const float* __restrict__ sw,
                                        float* __restrict__ X5, float* __restrict__ score,
                                        float* __restrict__ scOut, int R, int total)
{
    int i = blockIdx.x * blockDim.x + threadIdx.x;
    if (i >= total) return;
    int r = i % R;
    float sc = scale[r], sh = shift[r];
    float s = 0.0f;
    #pragma unroll
    for (int c = 0; c < 8; ++c) {
        float v = X[(long)i * 8 + c] * sc + sh;
        X5[(long)i * 8 + c] = v;
        s += v * sw[c];
    }
    float sg = 1.0f / (1.0f + expf(-s));
    score[i] = sg;
    scOut[i] = sg;
}

// ---------------------------------------------------------------------------
// Per-batch bitonic argsort (padded to 512), rank = inverse perm, gather top-L.
// ---------------------------------------------------------------------------
__global__ void __launch_bounds__(512)
sort_gather_kernel(const float* __restrict__ score, const float* __restrict__ X5,
                   float* __restrict__ h, int R, int L)
{
    __shared__ float key[512];
    __shared__ int   sidx[512];
    __shared__ int   rankA[512];
    int b = blockIdx.x, t = threadIdx.x;
    key[t]  = (t < R) ? score[(long)b * R + t] : 3.4e38f;
    sidx[t] = t;
    __syncthreads();
    for (int k = 2; k <= 512; k <<= 1) {
        for (int j = k >> 1; j > 0; j >>= 1) {
            int ixj = t ^ j;
            if (ixj > t) {
                bool up = ((t & k) == 0);
                bool gt = (key[t] > key[ixj]) ||
                          (key[t] == key[ixj] && sidx[t] > sidx[ixj]);
                if (gt == up) {
                    float tk = key[t]; key[t] = key[ixj]; key[ixj] = tk;
                    int   ti = sidx[t]; sidx[t] = sidx[ixj]; sidx[ixj] = ti;
                }
            }
            __syncthreads();
        }
    }
    if (sidx[t] < R) rankA[sidx[t]] = t;
    __syncthreads();
    if (t < L) {
        int src = rankA[t];
        const float* p = X5 + ((long)b * R + src) * 8;
        float* q = h + (long)b * (L * 8) + t * 8;
        float4 a = *(const float4*)p;
        float4 c = *(const float4*)(p + 4);
        *(float4*)q = a;
        *(float4*)(q + 4) = c;
    }
}

// out[b, :2] = H2[b, :32] @ Wf3[32,2] + bf3
__global__ void final_fc_kernel(const float* __restrict__ H2, const float* __restrict__ Wf3,
                                const float* __restrict__ bf3, float* __restrict__ out, int bz)
{
    int b = blockIdx.x * blockDim.x + threadIdx.x;
    if (b >= bz) return;
    float s0 = bf3[0], s1 = bf3[1];
    for (int k = 0; k < 32; ++k) {
        float v = H2[b * 32 + k];
        s0 += v * Wf3[k * 2 + 0];
        s1 += v * Wf3[k * 2 + 1];
    }
    out[b * 2 + 0] = s0;
    out[b * 2 + 1] = s1;
}

// ---------------------------------------------------------------------------
static void launch_gemm(int aff, int rsf, int leaky, dim3 grid, hipStream_t stream,
                        const float* A, const float* W, const float* bias,
                        const float* rowScale, const float* affS, const float* affB,
                        float* out, int M, int K, int N)
{
    dim3 blk(256);
    if (aff && rsf && leaky)
        gemm_wmma_kernel<true, true, true><<<grid, blk, 0, stream>>>(
            A, W, bias, rowScale, affS, affB, out, M, K, N);
    else if (!aff && rsf && leaky)
        gemm_wmma_kernel<false, true, true><<<grid, blk, 0, stream>>>(
            A, W, bias, rowScale, affS, affB, out, M, K, N);
    else if (!aff && !rsf && leaky)
        gemm_wmma_kernel<false, false, true><<<grid, blk, 0, stream>>>(
            A, W, bias, rowScale, affS, affB, out, M, K, N);
    else
        gemm_wmma_kernel<false, false, false><<<grid, blk, 0, stream>>>(
            A, W, bias, rowScale, affS, affB, out, M, K, N);
}

extern "C" void kernel_launch(void* const* d_in, const int* in_sizes, int n_in,
                              void* d_out, int out_size, void* d_ws, size_t ws_size,
                              hipStream_t stream)
{
    (void)in_sizes; (void)n_in; (void)out_size; (void)ws_size;
    const int bz = 256, R = 360, D = 360, L = 252;
    const int Mtot = bz * R;          // 92160

    const float* m   = (const float*)d_in[0];
    const float* NF  = (const float*)d_in[1];
    const float* W0  = (const float*)d_in[2];  const float* b0  = (const float*)d_in[3];
    const float* W1  = (const float*)d_in[4];  const float* b1  = (const float*)d_in[5];
    const float* g1  = (const float*)d_in[6];  const float* be1 = (const float*)d_in[7];
    const float* W2  = (const float*)d_in[8];  const float* b2  = (const float*)d_in[9];
    const float* g2  = (const float*)d_in[10]; const float* be2 = (const float*)d_in[11];
    const float* W3  = (const float*)d_in[12]; const float* b3  = (const float*)d_in[13];
    const float* W4  = (const float*)d_in[14]; const float* b4  = (const float*)d_in[15];
    const float* g3  = (const float*)d_in[16]; const float* be3 = (const float*)d_in[17];
    const float* sw  = (const float*)d_in[18];
    const float* Wf1 = (const float*)d_in[19]; const float* bf1 = (const float*)d_in[20];
    const float* Wf2 = (const float*)d_in[21]; const float* bf2 = (const float*)d_in[22];
    const float* Wf3 = (const float*)d_in[23]; const float* bf3 = (const float*)d_in[24];

    float* out = (float*)d_out;               // [bz,2] then sc [bz,R]
    float* scOut = out + bz * 2;

    float* ws = (float*)d_ws;
    size_t o = 0;
    float* rs    = ws + o; o += Mtot;
    float* stat  = ws + o; o += 2 * R;
    float* affS  = ws + o; o += R;
    float* affB  = ws + o; o += R;
    float* chS   = ws + o; o += R;
    float* chB   = ws + o; o += R;
    float* T4    = ws + o; o += (size_t)Mtot * 8;
    float* X5    = ws + o; o += (size_t)Mtot * 8;
    float* scBuf = ws + o; o += Mtot;
    float* hBuf  = ws + o; o += (size_t)bz * L * 8;
    float* H1    = ws + o; o += (size_t)bz * 256;
    float* H2    = ws + o; o += (size_t)bz * 32;
    float* bufA  = ws + o; o += (size_t)Mtot * D;
    float* bufB  = ws + o; o += (size_t)Mtot * D;

    dim3 gBig(Mtot / BM, (R + BN - 1) / BN);     // 720 x 6

    // rs = m.sum(-1)
    rowsum_kernel<<<(Mtot * 32 + 255) / 256, 256, 0, stream>>>(m, rs, R, Mtot);

    // T0 = leaky( (rs*NF) @ W0 + b0 )
    launch_gemm(0, 1, 1, gBig, stream, NF, W0, b0, rs, nullptr, nullptr,
                bufA, Mtot, D, R);
    // T1 = T0 @ W1 + b1
    launch_gemm(0, 0, 0, gBig, stream, bufA, W1, b1, nullptr, nullptr, nullptr,
                bufB, Mtot, R, R);
    // bn1 stats
    hipMemsetAsync(stat, 0, 2 * R * sizeof(float), stream);
    colstats_kernel<<<Mtot / 128, 384, 0, stream>>>(bufB, stat, stat + R, R, 128);
    bn_finalize_kernel<<<1, 512, 0, stream>>>(stat, stat + R, g1, be1, affS, affB,
                                              R, 1.0f / (float)Mtot);
    // T2 = leaky( (rs * bn1(T1)) @ W2 + b2 )
    launch_gemm(1, 1, 1, gBig, stream, bufB, W2, b2, rs, affS, affB,
                bufA, Mtot, R, R);
    // bn2 stats
    hipMemsetAsync(stat, 0, 2 * R * sizeof(float), stream);
    colstats_kernel<<<Mtot / 128, 384, 0, stream>>>(bufA, stat, stat + R, R, 128);
    bn_finalize_kernel<<<1, 512, 0, stream>>>(stat, stat + R, g2, be2, affS, affB,
                                              R, 1.0f / (float)Mtot);
    // T3 = leaky( (rs * bn2(T2)) @ W3 + b3 )   [Mtot, 64]
    launch_gemm(1, 1, 1, dim3(Mtot / BM, 1), stream, bufA, W3, b3, rs, affS, affB,
                bufB, Mtot, R, 64);
    // T4 = leaky( T3 @ W4 + b4 )   [Mtot, 8]
    fc64x8_kernel<<<Mtot / 256, 256, 0, stream>>>(bufB, W4, b4, T4, Mtot);

    // channel BN + score
    bnch_stats_kernel<<<R, 256, 0, stream>>>(T4, g3, be3, chS, chB, bz, R);
    bnch_apply_score_kernel<<<(Mtot + 255) / 256, 256, 0, stream>>>(
        T4, chS, chB, sw, X5, scBuf, scOut, R, Mtot);

    // argsort/rank/gather -> h [bz, L*8]
    sort_gather_kernel<<<bz, 512, 0, stream>>>(scBuf, X5, hBuf, R, L);

    // FC head (K = 2016 = 63*32 exactly)
    launch_gemm(0, 0, 1, dim3(bz / BM, 256 / BN), stream,
                hBuf, Wf1, bf1, nullptr, nullptr, nullptr, H1, bz, L * 8, 256);
    launch_gemm(0, 0, 1, dim3(bz / BM, 1), stream,
                H1, Wf2, bf2, nullptr, nullptr, nullptr, H2, bz, 256, 32);
    final_fc_kernel<<<1, 256, 0, stream>>>(H2, Wf3, bf3, out, bz);
}
